// META_66331474920018
// MI455X (gfx1250) — matrix operand order repaired and checked by reading, not compiled, not run on hardware
//
#include <hip/hip_runtime.h>
#include <hip/hip_bf16.h>

typedef __attribute__((ext_vector_type(16))) __bf16    v16bf;
typedef __attribute__((ext_vector_type(8)))  float     v8f;
typedef __attribute__((ext_vector_type(4)))  unsigned  uv4;

#define E_DIM  256
#define H_DIM  32
#define A_DIM  16
#define S_LEN  16
#define B_SZ   32768
#define H2_DIM 64
#define G4     128   // 4*H

__device__ __forceinline__ float fsigmoid(float x) { return 1.f / (1.f + __expf(-x)); }
__device__ __forceinline__ float ftanh(float x)    { float t = __expf(2.f * x); return (t - 1.f) / (t + 1.f); }

union HalfVec { uint4 q; __bf16 h[8]; };
union FragU   { uv4 q[2]; v16bf v; };

// Load 8 consecutive B-fragments (one N-tile row, stride 1024B) from LDS.
// asm volatile pins these in the loop body next to their WMMA consumers so
// LLVM cannot hoist them out of the time loop and spill 512 VGPRs to scratch.
// Ends with s_wait_dscnt 0 so the compiler's DS counter bookkeeping stays safe.
__device__ __forceinline__ void lds_load_brow(const v16bf* base, v16bf b[8]) {
    const unsigned off = (unsigned)(unsigned long long)base;  // low 32 bits = LDS byte addr
    uv4 q0, q1, q2, q3, q4, q5, q6, q7, q8, q9, qa, qb, qc, qd, qe, qf;
    asm volatile(
        "ds_load_b128 %0, %16\n\t"
        "ds_load_b128 %1, %16 offset:16\n\t"
        "ds_load_b128 %2, %16 offset:1024\n\t"
        "ds_load_b128 %3, %16 offset:1040\n\t"
        "ds_load_b128 %4, %16 offset:2048\n\t"
        "ds_load_b128 %5, %16 offset:2064\n\t"
        "ds_load_b128 %6, %16 offset:3072\n\t"
        "ds_load_b128 %7, %16 offset:3088\n\t"
        "ds_load_b128 %8, %16 offset:4096\n\t"
        "ds_load_b128 %9, %16 offset:4112\n\t"
        "ds_load_b128 %10, %16 offset:5120\n\t"
        "ds_load_b128 %11, %16 offset:5136\n\t"
        "ds_load_b128 %12, %16 offset:6144\n\t"
        "ds_load_b128 %13, %16 offset:6160\n\t"
        "ds_load_b128 %14, %16 offset:7168\n\t"
        "ds_load_b128 %15, %16 offset:7184\n\t"
        "s_wait_dscnt 0x0"
        : "=v"(q0), "=v"(q1), "=v"(q2), "=v"(q3),
          "=v"(q4), "=v"(q5), "=v"(q6), "=v"(q7),
          "=v"(q8), "=v"(q9), "=v"(qa), "=v"(qb),
          "=v"(qc), "=v"(qd), "=v"(qe), "=v"(qf)
        : "v"(off));
    FragU u;
    u.q[0] = q0; u.q[1] = q1; b[0] = u.v;
    u.q[0] = q2; u.q[1] = q3; b[1] = u.v;
    u.q[0] = q4; u.q[1] = q5; b[2] = u.v;
    u.q[0] = q6; u.q[1] = q7; b[3] = u.v;
    u.q[0] = q8; u.q[1] = q9; b[4] = u.v;
    u.q[0] = qa; u.q[1] = qb; b[5] = u.v;
    u.q[0] = qc; u.q[1] = qd; b[6] = u.v;
    u.q[0] = qe; u.q[1] = qf; b[7] = u.v;
}

// One block = (direction, 128-batch tile). 8 waves, each wave owns a 16-row M tile.
// Fused: embedding gather (bf16 A-frags) -> xg GEMM (WMMA bf16) -> recurrence
// (WMMA bf16, K=32 in one op) -> gates (f32 VALU) -> out[S,B,2H] (f32, d_ws).
__global__ __launch_bounds__(256) void lstm_dir_kernel(
    const float* __restrict__ emb,
    const float* __restrict__ Wih_f, const float* __restrict__ Whh_f,
    const float* __restrict__ bih_f, const float* __restrict__ bhh_f,
    const float* __restrict__ Wih_b, const float* __restrict__ Whh_b,
    const float* __restrict__ bih_b, const float* __restrict__ bhh_b,
    const int*   __restrict__ tokens,
    float*       __restrict__ outBuf)
{
    // B-matrix fragments (K x N): lane<16 holds N=lane, K=e; lane>=16 holds N=lane-16, K=16+e
    __shared__ v16bf sWih[8][8][32];   // [kt][nt][lane]  64 KB  (K=256 x N=128)
    __shared__ v16bf sWhh[8][32];      // [nt][lane]       8 KB  (K=32  x N=128)
    __shared__ float sBias[G4];
    __shared__ __align__(64) __bf16 sH[8][16][32];  // per-wave h tile (M=16 rows, N=32)

    const int tid  = threadIdx.x;
    const int bid  = blockIdx.x;
    const int dir  = bid >> 8;
    const int tile = bid & 255;

    const float* Wih = dir ? Wih_b : Wih_f;
    const float* Whh = dir ? Whh_b : Whh_f;
    const float* bih = dir ? bih_b : bih_f;
    const float* bhh = dir ? bhh_b : bhh_f;

    // ---- cooperative weight staging (f32 -> bf16, WMMA B-fragment layout) ----
    for (int i = tid; i < 8 * 8 * 32; i += 256) {
        const int kt = i >> 8, nt = (i >> 5) & 7, ln = i & 31;
        const int n  = nt * 16 + (ln & 15);
        const int kb = kt * 32 + (ln >> 4) * 16;
        v16bf v;
        #pragma unroll
        for (int e = 0; e < 16; ++e) v[e] = (__bf16)Wih[n * E_DIM + kb + e];
        sWih[kt][nt][ln] = v;
    }
    {
        const int nt = tid >> 5, ln = tid & 31;
        const int n  = nt * 16 + (ln & 15);
        const int kb = (ln >> 4) * 16;
        v16bf v;
        #pragma unroll
        for (int e = 0; e < 16; ++e) v[e] = (__bf16)Whh[n * H_DIM + kb + e];
        sWhh[nt][ln] = v;
    }
    if (tid < G4) sBias[tid] = bih[tid] + bhh[tid];
    __syncthreads();

    const int wave = tid >> 5;
    const int lane = tid & 31;
    const int m    = lane & 15;   // row within M tile (A) / N column (C,D)
    const int half = lane >> 4;
    const int b0   = tile * 128 + wave * 16;
    const int klo  = half * 16;   // A-frag K offset (contiguous: k = 16*half + e)

    // zero initial hidden state
    #pragma unroll
    for (int e = 0; e < 16; ++e) sH[wave][m][half * 16 + e] = (__bf16)0.f;

    v8f cst0 = {}, cst1 = {};    // cell state, D layout (16x32 tile = 2 N-tiles)
    const __bf16* hrow = &sH[wave][m][0];

    #pragma unroll 1
    for (int t = 0; t < S_LEN; ++t) {
        const int sIdx = dir ? (S_LEN - 1 - t) : t;
        const int tok  = tokens[(b0 + m) * S_LEN + sIdx];
        const float* erow = emb + (size_t)tok * E_DIM;

        // h as A-matrix fragment (16x32, K pattern {klo..klo+15})
        HalfVec u0, u1;
        u0.q = *(const uint4*)(hrow + klo);
        u1.q = *(const uint4*)(hrow + klo + 8);
        v16bf ha;
        #pragma unroll
        for (int e = 0; e < 8; ++e) { ha[e] = u0.h[e]; ha[8 + e] = u1.h[e]; }

        // recurrence part: acc = h @ Whh^T  (single K=32 WMMA per N-tile)
        v16bf hb[8];
        lds_load_brow(&sWhh[0][lane], hb);
        v8f acc[8];
        const v8f zero = {};
        #pragma unroll
        for (int nt = 0; nt < 8; ++nt)
            acc[nt] = __builtin_amdgcn_wmma_f32_16x16x32_bf16(
                false, ha, false, hb[nt], (short)0, zero, false, false);

        // input projection: acc += x @ Wih^T, gathering emb rows as bf16 A-frags
        #pragma unroll
        for (int kt = 0; kt < 8; ++kt) {
            const int base1 = kt * 32 + klo;
            const float4 f0 = *(const float4*)(erow + base1);
            const float4 f1 = *(const float4*)(erow + base1 + 4);
            const float4 f2 = *(const float4*)(erow + base1 + 8);
            const float4 f3 = *(const float4*)(erow + base1 + 12);
            v16bf xa;
            xa[0]  = (__bf16)f0.x; xa[1]  = (__bf16)f0.y; xa[2]  = (__bf16)f0.z; xa[3]  = (__bf16)f0.w;
            xa[4]  = (__bf16)f1.x; xa[5]  = (__bf16)f1.y; xa[6]  = (__bf16)f1.z; xa[7]  = (__bf16)f1.w;
            xa[8]  = (__bf16)f2.x; xa[9]  = (__bf16)f2.y; xa[10] = (__bf16)f2.z; xa[11] = (__bf16)f2.w;
            xa[12] = (__bf16)f3.x; xa[13] = (__bf16)f3.y; xa[14] = (__bf16)f3.z; xa[15] = (__bf16)f3.w;

            v16bf wb[8];
            lds_load_brow(&sWih[kt][0][lane], wb);
            #pragma unroll
            for (int nt = 0; nt < 8; ++nt)
                acc[nt] = __builtin_amdgcn_wmma_f32_16x16x32_bf16(
                    false, xa, false, wb[nt], (short)0, acc[nt], false, false);
        }

        // bias (n = nt*16 + m for every row r of the D tile)
        #pragma unroll
        for (int nt = 0; nt < 8; ++nt) {
            const float bn = sBias[nt * 16 + m];
            #pragma unroll
            for (int r = 0; r < 8; ++r) acc[nt][r] += bn;
        }

        // gates (torch order i,f,g,o): gate q occupies N-tiles {2q, 2q+1}
        const size_t obase = ((size_t)sIdx * B_SZ + b0) * H2_DIM + (size_t)dir * H_DIM;
        #pragma unroll
        for (int ct = 0; ct < 2; ++ct) {
            const v8f iv = acc[ct], fv = acc[2 + ct], gv = acc[4 + ct], ov = acc[6 + ct];
            v8f* cp = ct ? &cst1 : &cst0;
            #pragma unroll
            for (int r = 0; r < 8; ++r) {
                const float c = fsigmoid(fv[r]) * (*cp)[r] + fsigmoid(iv[r]) * ftanh(gv[r]);
                const float h = fsigmoid(ov[r]) * ftanh(c);
                (*cp)[r] = c;
                const int row = half ? (r + 8) : r;              // D layout: lane>=16 -> M=r+8
                sH[wave][row][ct * 16 + m] = (__bf16)h;          // stage for next step's A-frag
                outBuf[obase + (size_t)row * H2_DIM + ct * 16 + m] = h;
            }
        }
    }
}

// logit[k] = tanh(out_flat[k] @ w_omega) . u_omega ; exps[k] = exp(logit[k])
__global__ __launch_bounds__(256) void attn_logits_kernel(
    const float* __restrict__ w_omega, const float* __restrict__ u_omega,
    const float* __restrict__ outBuf,  float* __restrict__ exps)
{
    __shared__ float sW[H2_DIM][A_DIM];
    __shared__ float sU[A_DIM];
    const int tid = threadIdx.x;
    for (int i = tid; i < H2_DIM * A_DIM; i += 256) sW[i / A_DIM][i % A_DIM] = w_omega[i];
    if (tid < A_DIM) sU[tid] = u_omega[tid];
    __syncthreads();

    const int k = blockIdx.x * 256 + tid;
    const float* row = outBuf + (size_t)k * H2_DIM;
    float a[A_DIM];
    #pragma unroll
    for (int aa = 0; aa < A_DIM; ++aa) a[aa] = 0.f;
    #pragma unroll 4
    for (int j = 0; j < H2_DIM; ++j) {
        const float rj = row[j];
        #pragma unroll
        for (int aa = 0; aa < A_DIM; ++aa) a[aa] += rj * sW[j][aa];
    }
    float lg = 0.f;
    #pragma unroll
    for (int aa = 0; aa < A_DIM; ++aa) lg += ftanh(a[aa]) * sU[aa];
    exps[k] = __expf(lg);
}

// Softmax over groups of 16 consecutive flat rows (reference's reshape(-1,S)),
// then weighted sum over time: d_out[bb,ch] = sum_ss alpha[bb,ss]*out[ss,bb,ch]
__global__ __launch_bounds__(256) void attn_out_kernel(
    const float* __restrict__ exps, const float* __restrict__ outBuf,
    float* __restrict__ dout)
{
    const int gid = blockIdx.x * 256 + threadIdx.x;
    const int bb = gid >> 6, ch = gid & 63;
    float e[S_LEN], s = 0.f;
    #pragma unroll
    for (int ss = 0; ss < S_LEN; ++ss) { e[ss] = exps[bb * 16 + ss]; s += e[ss]; }
    const float inv = 1.f / s;
    float v = 0.f;
    #pragma unroll
    for (int ss = 0; ss < S_LEN; ++ss)
        v += e[ss] * outBuf[((size_t)ss * B_SZ + bb) * H2_DIM + ch];
    dout[(size_t)bb * H2_DIM + ch] = v * inv;
}

extern "C" void kernel_launch(void* const* d_in, const int* in_sizes, int n_in,
                              void* d_out, int out_size, void* d_ws, size_t ws_size,
                              hipStream_t stream)
{
    const float* emb   = (const float*)d_in[0];
    const float* Wih_f = (const float*)d_in[1];
    const float* Whh_f = (const float*)d_in[2];
    const float* bih_f = (const float*)d_in[3];
    const float* bhh_f = (const float*)d_in[4];
    const float* Wih_b = (const float*)d_in[5];
    const float* Whh_b = (const float*)d_in[6];
    const float* bih_b = (const float*)d_in[7];
    const float* bhh_b = (const float*)d_in[8];
    const float* w_om  = (const float*)d_in[9];
    const float* u_om  = (const float*)d_in[10];
    const int*   toks  = (const int*)d_in[11];
    (void)in_sizes; (void)n_in; (void)out_size; (void)ws_size;

    float* outBuf = (float*)d_ws;                                  // [S, B, 2H] f32 = 128 MB
    float* exps   = outBuf + (size_t)S_LEN * B_SZ * H2_DIM;        // [S*B] f32 = 2 MB

    lstm_dir_kernel<<<dim3(512), dim3(256), 0, stream>>>(
        emb, Wih_f, Whh_f, bih_f, bhh_f, Wih_b, Whh_b, bih_b, bhh_b, toks, outBuf);
    attn_logits_kernel<<<dim3((S_LEN * B_SZ) / 256), dim3(256), 0, stream>>>(
        w_om, u_om, outBuf, exps);
    attn_out_kernel<<<dim3((B_SZ * H2_DIM) / 256), dim3(256), 0, stream>>>(
        exps, outBuf, (float*)d_out);
}
